// BrainEncoder_32942399160808
// MI455X (gfx1250) — matrix operand order repaired and checked by reading, not compiled, not run on hardware
//
#include <hip/hip_runtime.h>
#include <math.h>

// ---------------- types ----------------
typedef __attribute__((ext_vector_type(16))) __bf16 v16bf;
typedef __attribute__((ext_vector_type(8)))  __bf16 v8bf;
typedef __attribute__((ext_vector_type(8)))  float  v8f;
typedef __attribute__((ext_vector_type(4)))  float  f32x4;

constexpr int kHidden = 256;
constexpr int kT      = 25;
constexpr int kTPad   = 27;     // +1 zero row each side for k=3 SAME conv
constexpr int kVid    = 768;
constexpr int kAud    = 128;
constexpr int kFrame  = 896;
constexpr int kSubj   = 4;
constexpr int kFmri   = 8192;
constexpr int kBatch  = 2048;
constexpr int kPos    = kBatch * kT;   // 51200, divisible by 32

static __device__ __forceinline__ v8f wmma_bf16(v16bf a, v16bf b, v8f c) {
  // D = A(16x32 bf16) * B(32x16 bf16) + C(16x16 f32)
  return __builtin_amdgcn_wmma_f32_16x16x32_bf16(false, a, false, b, (short)0, c,
                                                 false, false);
}
static __device__ __forceinline__ v16bf cat8(v8bf a, v8bf b) {
  return __builtin_shufflevector(a, b, 0, 1, 2, 3, 4, 5, 6, 7,
                                 8, 9, 10, 11, 12, 13, 14, 15);
}
static __device__ __forceinline__ float gelu_exact(float x) {
  return 0.5f * x * (1.0f + erff(x * 0.70710678118654752440f));
}

// Async global->LDS copy of 16 bytes (B128), tracked by ASYNCcnt.
// LDS operand is the low 32 bits of the generic shared-memory address
static __device__ __forceinline__ void async_copy_b128(void* lds_dst, const void* gsrc) {
  unsigned ldsoff = (unsigned)(unsigned long long)lds_dst;
  asm volatile("global_load_async_to_lds_b128 %0, %1, off"
               :
               : "v"(ldsoff), "v"(gsrc)
               : "memory");
}
static __device__ __forceinline__ void wait_async_all() {
  asm volatile("s_wait_asynccnt 0x0" ::: "memory");
}

// ---------------- prep kernels ----------------
// conv weight [Cout][Cin][3] -> tap-major bf16 hi/lo [3][Cout][Cin]
__global__ void prep_conv_w(const float* __restrict__ w, __bf16* __restrict__ wh,
                            __bf16* __restrict__ wl, int cin) {
  int i = blockIdx.x * 256 + threadIdx.x;
  int total = 3 * kHidden * cin;
  if (i >= total) return;
  int ci  = i % cin;
  int co  = (i / cin) % kHidden;
  int tap = i / (cin * kHidden);
  float v = w[((size_t)co * cin + ci) * 3 + tap];
  __bf16 h = (__bf16)v;
  wh[i] = h;
  wl[i] = (__bf16)(v - (float)h);
}

__global__ void prep_elem(const float* __restrict__ s, __bf16* __restrict__ h,
                          __bf16* __restrict__ l, int n) {
  int i = blockIdx.x * 256 + threadIdx.x;
  if (i >= n) return;
  float v = s[i];
  __bf16 hh = (__bf16)v;
  h[i] = hh;
  l[i] = (__bf16)(v - (float)hh);
}

// zero the t=0 and t=26 pad rows of both padded activation buffers (hi+lo)
__global__ void zero_pad_rows(__bf16* a, __bf16* b, __bf16* c, __bf16* d) {
  int i = blockIdx.x * 256 + threadIdx.x;
  if (i >= kBatch * 2 * kHidden) return;
  int ch = i % kHidden;
  int r  = (i / kHidden) & 1;
  int bb = i / (2 * kHidden);
  int tp = r ? (kTPad - 1) : 0;
  size_t off = ((size_t)bb * kTPad + tp) * kHidden + ch;
  __bf16 z = (__bf16)0.0f;
  a[off] = z; b[off] = z; c[off] = z; d[off] = z;
}

__global__ void zero_cnt(int* cnt) {
  if (threadIdx.x < kSubj) cnt[threadIdx.x] = 0;
}

__global__ void bucket_kernel(const int* __restrict__ sidx, int* cnt, int* perm) {
  int b = blockIdx.x * 256 + threadIdx.x;
  if (b >= kBatch) return;
  int s = sidx[b];
  int p = atomicAdd(&cnt[s], 1);
  perm[s * kBatch + p] = b;
}

// mean over T, emit bf16 hi/lo pooled features (deterministic, no atomics)
__global__ void pool_kernel(const float* __restrict__ y3, __bf16* __restrict__ ph,
                            __bf16* __restrict__ pl) {
  int i = blockIdx.x * 256 + threadIdx.x;
  if (i >= kBatch * kHidden) return;
  int c = i % kHidden;
  int b = i / kHidden;
  float s = 0.f;
  for (int t = 0; t < kT; ++t) s += y3[((size_t)(b * kT + t)) * kHidden + c];
  s *= (1.0f / (float)kT);
  __bf16 h = (__bf16)s;
  ph[i] = h;
  pl[i] = (__bf16)(s - (float)h);
}

// ---------------- fused conv(k=3) + GELU + LayerNorm ----------------
// Block tile: M = 256 out channels (8 waves x 2 M-tiles), N = 32 positions
// (2 N-tiles per wave -> A fragments reused across both, halving weight
// traffic), K = 3 taps x CIN in chunks of 32 (bf16 hi/lo split -> 3 WMMA per
// chunk per tile pair). Per tap, the 32-position activation slab (hi+lo) is
// staged once into LDS by the block (layers 2/3 use async global->LDS B128
// copies; layer 1 does the fp32->bf16 split during staging).
template <int CIN, int LAYER>
__global__ void __launch_bounds__(256)
conv_gelu_ln_kernel(const float* __restrict__ video, const float* __restrict__ audio,
                    const __bf16* __restrict__ xh, const __bf16* __restrict__ xl,
                    const __bf16* __restrict__ wh, const __bf16* __restrict__ wl,
                    const float* __restrict__ bias,
                    const float* __restrict__ gma, const float* __restrict__ bta,
                    __bf16* __restrict__ oh, __bf16* __restrict__ ol,
                    float* __restrict__ y3) {
  constexpr int kNPos = 32;
  // Staging slab (2 x 32 x CIN bf16) and LN tile (32 x 256 f32) live at
  // disjoint times -> union them.
  constexpr int kBfBytes   = 2 * kNPos * CIN * (int)sizeof(__bf16);
  constexpr int kTileBytes = kNPos * kHidden * (int)sizeof(float);
  constexpr int kSmemBytes = (kBfBytes > kTileBytes) ? kBfBytes : kTileBytes;
  __shared__ alignas(32) unsigned char smem[kSmemBytes];
  __shared__ float red_s[256];
  __shared__ float red_q[256];
  __shared__ float s_mu[kNPos];
  __shared__ float s_rs[kNPos];

  __bf16(*Xsh)[CIN] = (__bf16(*)[CIN])(smem);
  __bf16(*Xsl)[CIN] = (__bf16(*)[CIN])(smem + kNPos * CIN * sizeof(__bf16));
  float(*tile)[kHidden] = (float(*)[kHidden])(smem);

  const int tid  = threadIdx.x;
  const int lane = tid & 31;
  const int wave = tid >> 5;
  const int col  = lane & 15;
  const int half = lane >> 4;
  const int s8   = half * 8;
  const int s16  = half * 16;
  const int co0  = wave * 32;

  // staging / epilogue identity: 8 threads per position
  const int sp   = tid >> 3;                   // 0..31
  const int sj   = tid & 7;
  const int spos = blockIdx.x * kNPos + sp;
  const int sb   = spos / kT;
  const int st   = spos % kT;

  v8f acc00 = {0.f, 0.f, 0.f, 0.f, 0.f, 0.f, 0.f, 0.f};  // M-tile0, N-tile0
  v8f acc10 = acc00;                                      // M-tile1, N-tile0
  v8f acc01 = acc00;                                      // M-tile0, N-tile1
  v8f acc11 = acc00;                                      // M-tile1, N-tile1

  for (int tap = 0; tap < 3; ++tap) {
    // ---- stage activation slab for this tap into LDS ----
    if constexpr (LAYER == 1) {
      const int  ts  = st + tap - 1;
      const bool inb = (ts >= 0) && (ts < kT);
      for (int c = sj * 8; c < CIN; c += 64) {   // 14 chunks of 8 floats
        v8bf h8, l8;
        if (inb) {
          const float* src =
              (c < kVid)
                  ? (video + (size_t)sb * (kT * kVid) + (size_t)ts * kVid + c)
                  : (audio + (size_t)sb * (kT * kAud) + (size_t)ts * kAud + (c - kVid));
          const f32x4* s4 = (const f32x4*)src;
          f32x4 q0 = s4[0], q1 = s4[1];
          for (int i = 0; i < 4; ++i) {
            float v0 = q0[i], v1 = q1[i];
            __bf16 h0 = (__bf16)v0, h1 = (__bf16)v1;
            h8[i]     = h0;
            h8[4 + i] = h1;
            l8[i]     = (__bf16)(v0 - (float)h0);
            l8[4 + i] = (__bf16)(v1 - (float)h1);
          }
        } else {
          for (int i = 0; i < 8; ++i) { h8[i] = (__bf16)0.0f; l8[i] = (__bf16)0.0f; }
        }
        *(v8bf*)(&Xsh[sp][c]) = h8;
        *(v8bf*)(&Xsl[sp][c]) = l8;
      }
    } else {
      const size_t rowoff = ((size_t)(sb * kTPad + st + tap)) * CIN;
      const int    c      = sj * 32;             // 64B per thread per buffer
      for (int q = 0; q < 4; ++q) {
        async_copy_b128(&Xsh[sp][c + q * 8], xh + rowoff + c + q * 8);
        async_copy_b128(&Xsl[sp][c + q * 8], xl + rowoff + c + q * 8);
      }
      wait_async_all();
    }
    __syncthreads();

    const __bf16* w0h = wh + ((size_t)(tap * kHidden + co0 + col)) * CIN;
    const __bf16* w0l = wl + ((size_t)(tap * kHidden + co0 + col)) * CIN;
    const __bf16* w1h = w0h + (size_t)16 * CIN;
    const __bf16* w1l = w0l + (size_t)16 * CIN;

    for (int k0 = 0; k0 < CIN; k0 += 32) {
      if (k0 + 32 < CIN) {   // uniform branch; prefetch next weight chunk
        __builtin_prefetch((const void*)(w0h + k0 + 32), 0, 1);
        __builtin_prefetch((const void*)(w1h + k0 + 32), 0, 1);
      }
      v16bf b0h = *(const v16bf*)(&Xsh[col][k0 + s16]);
      v16bf b0l = *(const v16bf*)(&Xsl[col][k0 + s16]);
      v16bf b1h = *(const v16bf*)(&Xsh[16 + col][k0 + s16]);
      v16bf b1l = *(const v16bf*)(&Xsl[16 + col][k0 + s16]);

      v16bf a0h = *(const v16bf*)(w0h + k0 + s16);
      v16bf a0l = *(const v16bf*)(w0l + k0 + s16);
      v16bf a1h = *(const v16bf*)(w1h + k0 + s16);
      v16bf a1l = *(const v16bf*)(w1l + k0 + s16);

      acc00 = wmma_bf16(a0h, b0h, acc00);
      acc00 = wmma_bf16(a0h, b0l, acc00);
      acc00 = wmma_bf16(a0l, b0h, acc00);
      acc10 = wmma_bf16(a1h, b0h, acc10);
      acc10 = wmma_bf16(a1h, b0l, acc10);
      acc10 = wmma_bf16(a1l, b0h, acc10);
      acc01 = wmma_bf16(a0h, b1h, acc01);
      acc01 = wmma_bf16(a0h, b1l, acc01);
      acc01 = wmma_bf16(a0l, b1h, acc01);
      acc11 = wmma_bf16(a1h, b1h, acc11);
      acc11 = wmma_bf16(a1h, b1l, acc11);
      acc11 = wmma_bf16(a1l, b1h, acc11);
    }
    __syncthreads();   // slab is rewritten next tap / tile aliases the slab
  }

  // bias + exact GELU -> LDS tile[pos_local][channel]
  for (int r = 0; r < 8; ++r) {
    int m  = r + s8;           // D-layout row: r (+8 for lanes 16..31)
    int c0 = co0 + m;
    int c1 = co0 + 16 + m;
    tile[col][c0]      = gelu_exact(acc00[r] + bias[c0]);
    tile[col][c1]      = gelu_exact(acc10[r] + bias[c1]);
    tile[16 + col][c0] = gelu_exact(acc01[r] + bias[c0]);
    tile[16 + col][c1] = gelu_exact(acc11[r] + bias[c1]);
  }
  __syncthreads();

  // LayerNorm over 256 channels, 8 threads per position
  float sm = 0.f, sq = 0.f;
  for (int c = sj; c < kHidden; c += 8) {
    float v = tile[sp][c];
    sm += v;
    sq += v * v;
  }
  red_s[tid] = sm;
  red_q[tid] = sq;
  __syncthreads();
  if (sj == 0) {
    float a = 0.f, q = 0.f;
    for (int k = 0; k < 8; ++k) { a += red_s[sp * 8 + k]; q += red_q[sp * 8 + k]; }
    float mu  = a / (float)kHidden;
    float var = q / (float)kHidden - mu * mu;
    s_mu[sp] = mu;
    s_rs[sp] = rsqrtf(var + 1e-5f);
  }
  __syncthreads();

  const float mu = s_mu[sp];
  const float rs = s_rs[sp];
  for (int c = sj; c < kHidden; c += 8) {
    float v = (tile[sp][c] - mu) * rs * gma[c] + bta[c];
    if (y3) y3[((size_t)(sb * kT + st)) * kHidden + c] = v;
    if (oh) {
      size_t off = ((size_t)(sb * kTPad + st + 1)) * kHidden + c;
      __bf16 hh = (__bf16)v;
      oh[off] = hh;
      ol[off] = (__bf16)(v - (float)hh);
    }
  }
}

// ---------------- per-subject head GEMM ----------------
// M = fMRI rows (8 waves x 16), N = 32 same-subject batch columns (2 N-tiles
// per wave, A fragments reused), K = 256. Pooled features for the 32 columns
// are staged into LDS with async B128 copies.
__global__ void __launch_bounds__(256)
head_kernel(const __bf16* __restrict__ hwh, const __bf16* __restrict__ hwl,
            const float* __restrict__ hb,
            const __bf16* __restrict__ ph, const __bf16* __restrict__ pl,
            const int* __restrict__ cnt, const int* __restrict__ perm,
            float* __restrict__ out) {
  __shared__ alignas(32) __bf16 Psh[32][kHidden];
  __shared__ alignas(32) __bf16 Psl[32][kHidden];

  const int s     = blockIdx.z;
  const int btile = blockIdx.y;
  const int count = cnt[s];
  if (btile * 32 >= count) return;   // uniform: EXEC stays all-ones for WMMA

  const int tid  = threadIdx.x;
  const int lane = tid & 31;
  const int wave = tid >> 5;
  const int col  = lane & 15;
  const int half = lane >> 4;
  const int s8   = half * 8;
  const int s16  = half * 16;

  const int f0 = blockIdx.x * 128 + wave * 16;

  // stage 32 pooled rows (hi+lo) into LDS: 8 threads per row, 64B each/buffer
  {
    const int sp   = tid >> 3;
    const int sj   = tid & 7;
    const int ridx = btile * 32 + sp;
    const int srow = perm[s * kBatch + (ridx < count ? ridx : btile * 32)];
    const int c    = sj * 32;
    const __bf16* gh = ph + (size_t)srow * kHidden + c;
    const __bf16* gl = pl + (size_t)srow * kHidden + c;
    for (int q = 0; q < 4; ++q) {
      async_copy_b128(&Psh[sp][c + q * 8], gh + q * 8);
      async_copy_b128(&Psl[sp][c + q * 8], gl + q * 8);
    }
    wait_async_all();
  }
  __syncthreads();

  const int  idx0   = btile * 32 + col;
  const int  idx1   = btile * 32 + 16 + col;
  const bool valid0 = idx0 < count;
  const bool valid1 = idx1 < count;
  const int  bcol0  = perm[s * kBatch + (valid0 ? idx0 : btile * 32)];
  const int  bcol1  = perm[s * kBatch + (valid1 ? idx1 : btile * 32)];

  const __bf16* arh = hwh + ((size_t)(s * kFmri + f0 + col)) * kHidden;
  const __bf16* arl = hwl + ((size_t)(s * kFmri + f0 + col)) * kHidden;

  v8f acc0 = {0.f, 0.f, 0.f, 0.f, 0.f, 0.f, 0.f, 0.f};
  v8f acc1 = acc0;
  for (int k0 = 0; k0 < kHidden; k0 += 32) {
    v16bf ah = *(const v16bf*)(arh + k0 + s16);
    v16bf al = *(const v16bf*)(arl + k0 + s16);
    v16bf b0h = *(const v16bf*)(&Psh[col][k0 + s16]);
    v16bf b0l = *(const v16bf*)(&Psl[col][k0 + s16]);
    v16bf b1h = *(const v16bf*)(&Psh[16 + col][k0 + s16]);
    v16bf b1l = *(const v16bf*)(&Psl[16 + col][k0 + s16]);
    acc0 = wmma_bf16(ah, b0h, acc0);
    acc0 = wmma_bf16(ah, b0l, acc0);
    acc0 = wmma_bf16(al, b0h, acc0);
    acc1 = wmma_bf16(ah, b1h, acc1);
    acc1 = wmma_bf16(ah, b1l, acc1);
    acc1 = wmma_bf16(al, b1h, acc1);
  }

  const float* bro = hb + (size_t)s * kFmri + f0 + s8;
  if (valid0) {
    float* orow = out + (size_t)bcol0 * kFmri + f0 + s8;
    f32x4 lo4, hi4;
    for (int i = 0; i < 4; ++i) {
      lo4[i] = acc0[i] + bro[i];
      hi4[i] = acc0[4 + i] + bro[4 + i];
    }
    *(f32x4*)(orow)     = lo4;
    *(f32x4*)(orow + 4) = hi4;
  }
  if (valid1) {
    float* orow = out + (size_t)bcol1 * kFmri + f0 + s8;
    f32x4 lo4, hi4;
    for (int i = 0; i < 4; ++i) {
      lo4[i] = acc1[i] + bro[i];
      hi4[i] = acc1[4 + i] + bro[4 + i];
    }
    *(f32x4*)(orow)     = lo4;
    *(f32x4*)(orow + 4) = hi4;
  }
}

// ---------------- launch ----------------
extern "C" void kernel_launch(void* const* d_in, const int* in_sizes, int n_in,
                              void* d_out, int out_size, void* d_ws, size_t ws_size,
                              hipStream_t stream) {
  (void)in_sizes; (void)n_in; (void)out_size; (void)ws_size;
  const float* video = (const float*)d_in[0];
  const float* audio = (const float*)d_in[1];
  const int*   sidx  = (const int*)d_in[2];
  const float* c1w = (const float*)d_in[3];
  const float* c1b = (const float*)d_in[4];
  const float* c2w = (const float*)d_in[5];
  const float* c2b = (const float*)d_in[6];
  const float* c3w = (const float*)d_in[7];
  const float* c3b = (const float*)d_in[8];
  const float* g1  = (const float*)d_in[9];
  const float* be1 = (const float*)d_in[10];
  const float* g2  = (const float*)d_in[11];
  const float* be2 = (const float*)d_in[12];
  const float* g3  = (const float*)d_in[13];
  const float* be3 = (const float*)d_in[14];
  const float* hw  = (const float*)d_in[15];
  const float* hb  = (const float*)d_in[16];
  float* out = (float*)d_out;

  char* p = (char*)d_ws;
  auto take = [&](size_t bytes) -> char* {
    char* r = p;
    p += (bytes + 255) & ~(size_t)255;
    return r;
  };

  __bf16* w1h = (__bf16*)take((size_t)3 * kHidden * kFrame * sizeof(__bf16));
  __bf16* w1l = (__bf16*)take((size_t)3 * kHidden * kFrame * sizeof(__bf16));
  __bf16* w2h = (__bf16*)take((size_t)3 * kHidden * kHidden * sizeof(__bf16));
  __bf16* w2l = (__bf16*)take((size_t)3 * kHidden * kHidden * sizeof(__bf16));
  __bf16* w3h = (__bf16*)take((size_t)3 * kHidden * kHidden * sizeof(__bf16));
  __bf16* w3l = (__bf16*)take((size_t)3 * kHidden * kHidden * sizeof(__bf16));
  __bf16* hwh = (__bf16*)take((size_t)kSubj * kFmri * kHidden * sizeof(__bf16));
  __bf16* hwl = (__bf16*)take((size_t)kSubj * kFmri * kHidden * sizeof(__bf16));
  __bf16* x1h = (__bf16*)take((size_t)kBatch * kTPad * kHidden * sizeof(__bf16));
  __bf16* x1l = (__bf16*)take((size_t)kBatch * kTPad * kHidden * sizeof(__bf16));
  __bf16* x2h = (__bf16*)take((size_t)kBatch * kTPad * kHidden * sizeof(__bf16));
  __bf16* x2l = (__bf16*)take((size_t)kBatch * kTPad * kHidden * sizeof(__bf16));
  float*  y3  = (float*) take((size_t)kBatch * kT * kHidden * sizeof(float));
  __bf16* ph  = (__bf16*)take((size_t)kBatch * kHidden * sizeof(__bf16));
  __bf16* pl  = (__bf16*)take((size_t)kBatch * kHidden * sizeof(__bf16));
  int*    cnt = (int*)   take(kSubj * sizeof(int));
  int*    perm= (int*)   take((size_t)kSubj * kBatch * sizeof(int));

  // weight repack to bf16 hi/lo
  {
    int n1 = 3 * kHidden * kFrame;
    prep_conv_w<<<(n1 + 255) / 256, 256, 0, stream>>>(c1w, w1h, w1l, kFrame);
    int n2 = 3 * kHidden * kHidden;
    prep_conv_w<<<(n2 + 255) / 256, 256, 0, stream>>>(c2w, w2h, w2l, kHidden);
    prep_conv_w<<<(n2 + 255) / 256, 256, 0, stream>>>(c3w, w3h, w3l, kHidden);
    int nh = kSubj * kFmri * kHidden;
    prep_elem<<<(nh + 255) / 256, 256, 0, stream>>>(hw, hwh, hwl, nh);
  }
  zero_pad_rows<<<(kBatch * 2 * kHidden + 255) / 256, 256, 0, stream>>>(x1h, x1l, x2h, x2l);
  zero_cnt<<<1, 32, 0, stream>>>(cnt);
  bucket_kernel<<<(kBatch + 255) / 256, 256, 0, stream>>>(sidx, cnt, perm);

  // three fused conv+GELU+LN stages (32 positions per block)
  conv_gelu_ln_kernel<kFrame, 1><<<kPos / 32, 256, 0, stream>>>(
      video, audio, nullptr, nullptr, w1h, w1l, c1b, g1, be1, x1h, x1l, nullptr);
  conv_gelu_ln_kernel<kHidden, 2><<<kPos / 32, 256, 0, stream>>>(
      nullptr, nullptr, x1h, x1l, w2h, w2l, c2b, g2, be2, x2h, x2l, nullptr);
  conv_gelu_ln_kernel<kHidden, 2><<<kPos / 32, 256, 0, stream>>>(
      nullptr, nullptr, x2h, x2l, w3h, w3l, c3b, g3, be3, nullptr, nullptr, y3);

  // temporal mean pool, then per-subject head GEMM (32 columns per block)
  pool_kernel<<<(kBatch * kHidden + 255) / 256, 256, 0, stream>>>(y3, ph, pl);
  head_kernel<<<dim3(kFmri / 128, kBatch / 32, kSubj), 256, 0, stream>>>(
      hwh, hwl, hb, ph, pl, cnt, perm, out);
}
